// CARTE_Attention_47983374631141
// MI455X (gfx1250) — matrix operand order repaired and checked by reading, not compiled, not run on hardware
//
#include <hip/hip_runtime.h>
#include <hip/hip_bf16.h>

typedef __attribute__((ext_vector_type(16))) __bf16 v16bf;
typedef __attribute__((ext_vector_type(8)))  float  v8f;

// ---------- helpers ----------
__device__ __forceinline__ unsigned short f2bf(float f) {
    union { float f; unsigned u; } a; a.f = f;
    unsigned u = a.u;
    u += 0x7fffu + ((u >> 16) & 1u);   // round-to-nearest-even
    return (unsigned short)(u >> 16);
}
__device__ __forceinline__ float bf2f(unsigned short s) {
    union { unsigned u; float f; } a; a.u = ((unsigned)s) << 16; return a.f;
}
// monotone float -> uint mapping (atomicMax on uint == float max). init 0 == below -inf.
__device__ __forceinline__ unsigned f2ord(float f) {
    unsigned u = __float_as_uint(f);
    return (u & 0x80000000u) ? ~u : (u | 0x80000000u);
}
__device__ __forceinline__ float ord2f(unsigned k) {
    unsigned u = (k & 0x80000000u) ? (k & 0x7fffffffu) : ~k;
    return __uint_as_float(u);
}
__device__ __forceinline__ unsigned pack2bf(float a, float b) {
    return (unsigned)f2bf(a) | ((unsigned)f2bf(b) << 16);
}

// A-fragment (16-bit, 16x32): lane half 0 holds K {0-7,16-23}, half 1 holds K {8-15,24-31}
// -> two contiguous 8-ushort (16B) chunks per lane: guaranteed ds_load_b128.
__device__ __forceinline__ v16bf load_afrag(const unsigned short* row, int kk, int half) {
    union { uint4 q[2]; v16bf v; } u;
    u.q[0] = *(const uint4*)(row + 32 * kk + 8 * half);
    u.q[1] = *(const uint4*)(row + 32 * kk + 16 + 8 * half);
    return u.v;
}
// B-fragment: weights pre-packed so each lane's 16 bf16 are contiguous (32B aligned).
__device__ __forceinline__ v16bf load_bfrag(const unsigned short* Wp, int nt, int kk, int lane) {
    union { uint4 q[2]; v16bf v; } u;
    const uint4* p = (const uint4*)(Wp + ((((nt << 2) + kk) << 5) + lane) * 16);
    u.q[0] = p[0];
    u.q[1] = p[1];
    return u.v;
}

// ---------- K0: init accumulators ----------
__global__ void k_init(float* out_acc, unsigned* segmax, float* denom, int nOut, int nSeg) {
    int i = blockIdx.x * blockDim.x + threadIdx.x;
    if (i < nOut) out_acc[i] = 0.0f;
    if (i < nSeg) { segmax[i] = 0u; denom[i] = 0.0f; }
}

// ---------- K1a: convert + pack weights into B-fragment order ----------
// Packed layout per matrix: [n_tile(8)][kk(4)][lane(32)][e(16)] bf16, 16384 elements.
// Source element: k = 32*kk + 8*(lane>=16) + e + 8*(e>=8),  n = n_tile*16 + (lane&15).
__global__ void k_cvt_w(const float* Wq, const float* Wk, const float* Wv, const float* We,
                        unsigned short* o) {
    int i = blockIdx.x * blockDim.x + threadIdx.x;
    if (i >= 4 * 16384) return;
    int m = i >> 14, p = i & 16383;
    const float* W = (m == 0) ? Wq : (m == 1) ? Wk : (m == 2) ? Wv : We;
    int e    = p & 15;
    int lane = (p >> 4) & 31;
    int kk   = (p >> 9) & 3;
    int nt   = (p >> 11) & 7;
    int k = 32 * kk + 8 * (lane >> 4) + e + 8 * (e >> 3);
    int n = nt * 16 + (lane & 15);
    o[i] = f2bf(W[k * 128 + n]);
}

// ---------- K1: q = x @ Wq  (bf16 WMMA, f32 accumulate) ----------
__global__ void __launch_bounds__(256) k_qgemm(const float* __restrict__ x,
                                               const unsigned short* __restrict__ Wqp,
                                               float* __restrict__ q) {
    __shared__ unsigned short lA[16][128];
    const int row0 = blockIdx.x * 16;
    for (int idx = threadIdx.x; idx < 16 * 32; idx += 256) {
        int r = idx >> 5, c = (idx & 31) << 2;
        float4 xv = *(const float4*)(x + (size_t)(row0 + r) * 128 + c);
        uint2 pk;
        pk.x = pack2bf(xv.x, xv.y);
        pk.y = pack2bf(xv.z, xv.w);
        *(uint2*)&lA[r][c] = pk;
    }
    __syncthreads();

    const int lane = threadIdx.x & 31;
    const int half = lane >> 4;
    const int l16  = lane & 15;
    const int nt   = threadIdx.x >> 5;   // wave -> column tile

    v8f acc = {};
#pragma unroll
    for (int kk = 0; kk < 4; ++kk) {
        v16bf a = load_afrag(&lA[l16][0], kk, half);
        v16bf b = load_bfrag(Wqp, nt, kk, lane);
        acc = __builtin_amdgcn_wmma_f32_16x16x32_bf16(false, a, false, b, (short)0, acc,
                                                      false, false);
    }
#pragma unroll
    for (int j = 0; j < 8; ++j) {
        int m = j + 8 * half;
        q[(size_t)(row0 + m) * 128 + nt * 16 + l16] = acc[j];
    }
}

// ---------- K2: fused per-edge  Z = ea*x[e1];  k=Z@Wk; v=Z@Wv; eo=ea@We+be; scores ----------
__global__ void __launch_bounds__(256) k_edge(const float* __restrict__ x,
                                              const float* __restrict__ edge_attr,
                                              const int* __restrict__ e0arr,
                                              const int* __restrict__ e1arr,
                                              const unsigned short* __restrict__ Wp,
                                              const float* __restrict__ q,
                                              const float* __restrict__ be,
                                              float* __restrict__ score,
                                              unsigned* __restrict__ segmax,
                                              unsigned short* __restrict__ vbf,
                                              float* __restrict__ edge_out) {
    __shared__ unsigned short lZ[16][128];
    __shared__ unsigned short lE[16][128];
    __shared__ float kt[8][16][20];      // stride 20 floats: 16B aligned rows, pads banks
    __shared__ int se0[16], se1[16];

    const int eb = blockIdx.x * 16;
    if (threadIdx.x < 16) {
        se0[threadIdx.x] = e0arr[eb + threadIdx.x];
        se1[threadIdx.x] = e1arr[eb + threadIdx.x];
    }
    __syncthreads();

    for (int idx = threadIdx.x; idx < 16 * 32; idx += 256) {
        int r = idx >> 5, c = (idx & 31) << 2;
        float4 ea = *(const float4*)(edge_attr + (size_t)(eb + r) * 128 + c);
        float4 xv = *(const float4*)(x + (size_t)se1[r] * 128 + c);
        uint2 pe, pz;
        pe.x = pack2bf(ea.x, ea.y);
        pe.y = pack2bf(ea.z, ea.w);
        pz.x = pack2bf(ea.x * xv.x, ea.y * xv.y);
        pz.y = pack2bf(ea.z * xv.z, ea.w * xv.w);
        *(uint2*)&lE[r][c] = pe;
        *(uint2*)&lZ[r][c] = pz;
    }
    __syncthreads();

    const int lane = threadIdx.x & 31;
    const int half = lane >> 4;
    const int l16  = lane & 15;
    const int nt   = threadIdx.x >> 5;   // wave -> column tile == head

    const unsigned short* Wkp = Wp + 16384;
    const unsigned short* Wvp = Wp + 2 * 16384;
    const unsigned short* Wep = Wp + 3 * 16384;

    v8f ak = {}, av = {}, ae = {};
#pragma unroll
    for (int kk = 0; kk < 4; ++kk) {
        v16bf a  = load_afrag(&lZ[l16][0], kk, half);
        v16bf aE = load_afrag(&lE[l16][0], kk, half);
        v16bf bk = load_bfrag(Wkp, nt, kk, lane);
        v16bf bv = load_bfrag(Wvp, nt, kk, lane);
        v16bf bE = load_bfrag(Wep, nt, kk, lane);
        ak = __builtin_amdgcn_wmma_f32_16x16x32_bf16(false, a,  false, bk, (short)0, ak, false, false);
        av = __builtin_amdgcn_wmma_f32_16x16x32_bf16(false, a,  false, bv, (short)0, av, false, false);
        ae = __builtin_amdgcn_wmma_f32_16x16x32_bf16(false, aE, false, bE, (short)0, ae, false, false);
    }

    const float bias = be[nt * 16 + l16];
#pragma unroll
    for (int j = 0; j < 8; ++j) {
        int m = j + 8 * half;
        kt[nt][m][l16] = ak[j];                                        // head nt's k-tile
        vbf[(size_t)(eb + m) * 128 + nt * 16 + l16] = f2bf(av[j]);     // v spilled as bf16
        edge_out[(size_t)(eb + m) * 128 + nt * 16 + l16] = ae[j] + bias;
    }
    __syncthreads();

    if (threadIdx.x < 128) {
        int h  = threadIdx.x >> 4;
        int el = threadIdx.x & 15;
        int dst = se0[el];
        const float4* qv = (const float4*)(q + (size_t)dst * 128 + h * 16);
        const float4* kv = (const float4*)&kt[h][el][0];
        float s = 0.0f;
#pragma unroll
        for (int c4 = 0; c4 < 4; ++c4) {
            float4 qq = qv[c4], kk4 = kv[c4];
            s += qq.x * kk4.x + qq.y * kk4.y + qq.z * kk4.z + qq.w * kk4.w;
        }
        s *= 0.25f;                                   // 1/sqrt(16)
        score[(size_t)(eb + el) * 8 + h] = s;
        atomicMax(&segmax[dst * 8 + h], f2ord(s));
    }
}

// ---------- K3: alpha = exp(score - max); denom += alpha; out += alpha*v ----------
__global__ void k_soft(const float* __restrict__ score, const int* __restrict__ e0arr,
                       const unsigned* __restrict__ segmax, const unsigned short* __restrict__ vbf,
                       float* __restrict__ denom, float* __restrict__ out_acc, int total) {
    int i = blockIdx.x * blockDim.x + threadIdx.x;
    if (i >= total) return;
    int e = i >> 3, h = i & 7;
    int dst = e0arr[e];
    float m = ord2f(segmax[dst * 8 + h]);
    float alpha = __expf(score[i] - m);
    atomicAdd(&denom[dst * 8 + h], alpha);
    union { uint4 q[2]; unsigned short s[16]; } vu;
    const uint4* vp = (const uint4*)(vbf + (size_t)e * 128 + h * 16);
    vu.q[0] = vp[0];
    vu.q[1] = vp[1];
    float* op = out_acc + (size_t)dst * 128 + h * 16;
#pragma unroll
    for (int c = 0; c < 16; ++c) atomicAdd(&op[c], alpha * bf2f(vu.s[c]));
}

// ---------- K4: normalize ----------
__global__ void k_norm(float* out_acc, const float* __restrict__ denom, int total4) {
    int i = blockIdx.x * blockDim.x + threadIdx.x;
    if (i >= total4) return;
    int n = i >> 5, c = (i & 31) << 2, h = c >> 4;
    float d = 1.0f / denom[n * 8 + h];
    float4* p = (float4*)(out_acc + (size_t)n * 128 + c);
    float4 v = *p;
    v.x *= d; v.y *= d; v.z *= d; v.w *= d;
    *p = v;
}

// ---------- launch ----------
extern "C" void kernel_launch(void* const* d_in, const int* in_sizes, int n_in,
                              void* d_out, int out_size, void* d_ws, size_t ws_size,
                              hipStream_t stream) {
    const float* x         = (const float*)d_in[0];
    const float* edge_attr = (const float*)d_in[1];
    const float* Wq        = (const float*)d_in[2];
    const float* Wk        = (const float*)d_in[3];
    const float* Wv        = (const float*)d_in[4];
    const float* We        = (const float*)d_in[5];
    const float* be        = (const float*)d_in[6];
    const int*   eidx      = (const int*)d_in[7];

    const int N = in_sizes[0] / 128;   // 65536
    const int E = in_sizes[7] / 2;     // 400000
    const int* e0 = eidx;
    const int* e1 = eidx + E;

    float* out      = (float*)d_out;          // [N,128]
    float* edge_out = out + (size_t)N * 128;  // [E,128]

    // workspace layout
    char* w = (char*)d_ws;
    unsigned short* Wb   = (unsigned short*)w; w += (size_t)4 * 16384 * sizeof(unsigned short);
    float* qf            = (float*)w;          w += (size_t)N * 128 * sizeof(float);
    float* score         = (float*)w;          w += (size_t)E * 8 * sizeof(float);
    unsigned* segmax     = (unsigned*)w;       w += (size_t)N * 8 * sizeof(unsigned);
    float* denom         = (float*)w;          w += (size_t)N * 8 * sizeof(float);
    unsigned short* vbf  = (unsigned short*)w; w += (size_t)E * 128 * sizeof(unsigned short);
    (void)ws_size; (void)n_in; (void)out_size;

    const int nOut = N * 128, nSeg = N * 8;
    k_init <<<(nOut + 255) / 256, 256, 0, stream>>>(out, segmax, denom, nOut, nSeg);
    k_cvt_w<<<(4 * 16384 + 255) / 256, 256, 0, stream>>>(Wq, Wk, Wv, We, Wb);
    k_qgemm<<<N / 16, 256, 0, stream>>>(x, Wb, qf);
    k_edge <<<E / 16, 256, 0, stream>>>(x, edge_attr, e0, e1, Wb, qf, be,
                                        score, segmax, vbf, edge_out);
    k_soft <<<(E * 8 + 255) / 256, 256, 0, stream>>>(score, e0, segmax, vbf, denom, out, E * 8);
    k_norm <<<(N * 32 + 255) / 256, 256, 0, stream>>>(out, denom, N * 32);
}
